// BinaryXnorExceptOutliersLinear_53824530153497
// MI455X (gfx1250) — compile-verified, matrix operand order checked by reading, not run on hardware
//
#include <hip/hip_runtime.h>
#include <hip/hip_bf16.h>

// ---------- types ----------
typedef __attribute__((ext_vector_type(16))) __bf16 v16bf;
typedef __attribute__((ext_vector_type(8)))  __bf16 v8bf;
typedef __attribute__((ext_vector_type(4)))  __bf16 v4bf;
typedef __attribute__((ext_vector_type(8)))  float  v8f;

#define M_DIM 8192
#define N_DIM 4096
#define K_DIM 4096
#define NW    (4096 * 4096)

#define RED_BLOCKS 1024
#define RED_THREADS 256

// d_ws layout:
//  doubles [0..4095]: reduction partials
//  float region at (float*)(wsd + 4096): f[0]=lower f[1]=upper f[2]=scale
//  byte offset 1MB:        w_bin as bf16   [4096 x 4096]  (32 MB)
//  byte offset 1MB+32MB:   x as bf16       [8192 x 4096]  (64 MB)
#define WS_F_OFF 4096

// ============================================================
// Stats passes (deterministic tree reductions)
// ============================================================
__global__ __launch_bounds__(RED_THREADS)
void stats_partial1(const float* __restrict__ w, double* __restrict__ wsd) {
    __shared__ double s1[RED_THREADS];
    __shared__ double s2[RED_THREADS];
    double a = 0.0, b = 0.0;
    for (int i = blockIdx.x * blockDim.x + threadIdx.x; i < NW;
         i += gridDim.x * blockDim.x) {
        double v = (double)w[i];
        a += v;
        b += v * v;
    }
    s1[threadIdx.x] = a;
    s2[threadIdx.x] = b;
    __syncthreads();
    for (int s = RED_THREADS / 2; s > 0; s >>= 1) {
        if (threadIdx.x < s) {
            s1[threadIdx.x] += s1[threadIdx.x + s];
            s2[threadIdx.x] += s2[threadIdx.x + s];
        }
        __syncthreads();
    }
    if (threadIdx.x == 0) {
        wsd[blockIdx.x]        = s1[0];
        wsd[1024 + blockIdx.x] = s2[0];
    }
}

__global__ __launch_bounds__(RED_THREADS)
void stats_finalize1(double* __restrict__ wsd) {
    __shared__ double s1[RED_THREADS];
    __shared__ double s2[RED_THREADS];
    double a = 0.0, b = 0.0;
    for (int i = threadIdx.x; i < RED_BLOCKS; i += RED_THREADS) {
        a += wsd[i];
        b += wsd[1024 + i];
    }
    s1[threadIdx.x] = a;
    s2[threadIdx.x] = b;
    __syncthreads();
    for (int s = RED_THREADS / 2; s > 0; s >>= 1) {
        if (threadIdx.x < s) {
            s1[threadIdx.x] += s1[threadIdx.x + s];
            s2[threadIdx.x] += s2[threadIdx.x + s];
        }
        __syncthreads();
    }
    if (threadIdx.x == 0) {
        const double n    = (double)NW;
        const double mean = s1[0] / n;
        const double var  = (s2[0] - n * mean * mean) / (n - 1.0);  // ddof=1
        const double sd   = sqrt(var);
        float* f = (float*)(wsd + WS_F_OFF);
        f[0] = (float)(mean - 1.96 * sd);
        f[1] = (float)(mean + 1.96 * sd);
    }
}

__global__ __launch_bounds__(RED_THREADS)
void stats_partial2(const float* __restrict__ w, double* __restrict__ wsd) {
    __shared__ double s1[RED_THREADS];
    __shared__ double s2[RED_THREADS];
    const float* f = (const float*)(wsd + WS_F_OFF);
    const float lo = f[0];
    const float hi = f[1];
    double a = 0.0, b = 0.0;
    for (int i = blockIdx.x * blockDim.x + threadIdx.x; i < NW;
         i += gridDim.x * blockDim.x) {
        float v = w[i];
        if (!(v < lo || v > hi)) {
            a += (double)fabsf(v);
            b += 1.0;
        }
    }
    s1[threadIdx.x] = a;
    s2[threadIdx.x] = b;
    __syncthreads();
    for (int s = RED_THREADS / 2; s > 0; s >>= 1) {
        if (threadIdx.x < s) {
            s1[threadIdx.x] += s1[threadIdx.x + s];
            s2[threadIdx.x] += s2[threadIdx.x + s];
        }
        __syncthreads();
    }
    if (threadIdx.x == 0) {
        wsd[2048 + blockIdx.x] = s1[0];
        wsd[3072 + blockIdx.x] = s2[0];
    }
}

__global__ __launch_bounds__(RED_THREADS)
void stats_finalize2(double* __restrict__ wsd) {
    __shared__ double s1[RED_THREADS];
    __shared__ double s2[RED_THREADS];
    double a = 0.0, b = 0.0;
    for (int i = threadIdx.x; i < RED_BLOCKS; i += RED_THREADS) {
        a += wsd[2048 + i];
        b += wsd[3072 + i];
    }
    s1[threadIdx.x] = a;
    s2[threadIdx.x] = b;
    __syncthreads();
    for (int s = RED_THREADS / 2; s > 0; s >>= 1) {
        if (threadIdx.x < s) {
            s1[threadIdx.x] += s1[threadIdx.x + s];
            s2[threadIdx.x] += s2[threadIdx.x + s];
        }
        __syncthreads();
    }
    if (threadIdx.x == 0) {
        float* f = (float*)(wsd + WS_F_OFF);
        f[2] = (float)(s1[0] / s2[0]);
    }
}

// ============================================================
// Pre-convert passes: done ONCE instead of per GEMM block.
// ============================================================
__global__ __launch_bounds__(256)
void cvt_x_bf16(const float* __restrict__ X, __bf16* __restrict__ xb) {
    const size_t i4 = (size_t)blockIdx.x * 256 + threadIdx.x;  // float4 index
    const float4 v = ((const float4*)X)[i4];
    v4bf p;
    p[0] = (__bf16)v.x;
    p[1] = (__bf16)v.y;
    p[2] = (__bf16)v.z;
    p[3] = (__bf16)v.w;
    ((v4bf*)xb)[i4] = p;
}

__global__ __launch_bounds__(256)
void cvt_w_bf16(const float* __restrict__ W, __bf16* __restrict__ wb,
                const float* __restrict__ st) {
    const float lo = st[0], hi = st[1], scale = st[2];
    const size_t i4 = (size_t)blockIdx.x * 256 + threadIdx.x;
    const float4 v = ((const float4*)W)[i4];
    const float b0 = (v.x < lo || v.x > hi) ? v.x : v.x * scale;
    const float b1 = (v.y < lo || v.y > hi) ? v.y : v.y * scale;
    const float b2 = (v.z < lo || v.z > hi) ? v.z : v.z * scale;
    const float b3 = (v.w < lo || v.w > hi) ? v.w : v.w * scale;
    v4bf p;
    p[0] = (__bf16)b0;
    p[1] = (__bf16)b1;
    p[2] = (__bf16)b2;
    p[3] = (__bf16)b3;
    ((v4bf*)wb)[i4] = p;
}

// ============================================================
// Shared GEMM pieces
// ============================================================
#define BK  32
#define LDK 40   // padded LDS row stride in bf16 elements (80B, 16B-aligned)
#define BUFE (128 * LDK)      // elements per LDS buffer (one matrix)
#define BUFB (BUFE * 2)       // bytes per LDS buffer

__device__ __forceinline__ v16bf load_frag(const __bf16* __restrict__ base, int lane) {
    const int r  = lane & 15;
    const int kh = (lane >> 4) << 3;  // 0 or 8
    const __bf16* p = base + r * LDK;
    v8bf lo = *(const v8bf*)(p + kh);
    v8bf hi = *(const v8bf*)(p + 16 + kh);
    v16bf f;
#pragma unroll
    for (int i = 0; i < 8; ++i) {
        f[i]     = lo[i];
        f[i + 8] = hi[i];
    }
    return f;
}

// Async 16B copy global(bf16) -> LDS via the CDNA5 async engine (ASYNCcnt).
__device__ __forceinline__ void async_copy_b128(unsigned lds_addr,
                                                const void* base,
                                                unsigned voff) {
    asm volatile("global_load_async_to_lds_b128 %0, %1, %2"
                 :
                 : "v"(lds_addr), "v"(voff), "s"(base)
                 : "memory");
}
__device__ __forceinline__ void wait_async_le4() {
    asm volatile("s_wait_asynccnt 0x4" ::: "memory");
}
__device__ __forceinline__ void wait_async_0() {
    asm volatile("s_wait_asynccnt 0x0" ::: "memory");
}

// ============================================================
// GEMM (async path): bf16 operands pre-staged in ws.
// 8 waves; tile 128x128, BK=32; TRIPLE-buffered LDS; copies run
// 2 stages ahead of compute; in-order ASYNCcnt lets us wait only
// on the older copy group (s_wait_asynccnt 4).
// ============================================================
__global__ __launch_bounds__(256)
void gemm_bf16_async(const __bf16* __restrict__ XB, const __bf16* __restrict__ WB,
                     const float* __restrict__ bias, float* __restrict__ out) {
    __shared__ __bf16 As[3][BUFE];  // 3 x 10 KB
    __shared__ __bf16 Bs[3][BUFE];  // 3 x 10 KB

    const int tid  = threadIdx.x;
    const int lane = tid & 31;
    const int wave = tid >> 5;
    const int wm   = wave >> 1;
    const int wn   = wave & 1;

    const int nBase = blockIdx.x * 128;
    const int mBase = blockIdx.y * 128;

    // Copier mapping: chunk c = 16B (8 bf16); 4 chunks/row; rows r0, r0+64.
    const int c  = tid & 3;
    const int r0 = tid >> 2;  // 0..63
    const unsigned rowBytes = K_DIM * 2;  // 8192
    const unsigned ldsRow   = r0 * (LDK * 2) + c * 16u;
    const unsigned ldsA     = (unsigned)(uintptr_t)&As[0][0];
    const unsigned ldsB     = (unsigned)(uintptr_t)&Bs[0][0];
    const unsigned rowHop   = 64 * (LDK * 2);  // +64 rows in LDS
    const unsigned gRowHop  = 64u * rowBytes;  // +64 rows in global

    // Running global byte offsets of the NEXT stage to issue.
    unsigned aOff = (unsigned)((mBase + r0) * (size_t)rowBytes) + c * 16u;
    unsigned bOff = (unsigned)((nBase + r0) * (size_t)rowBytes) + c * 16u;

    auto issue_stage = [&](int buf) {
        const unsigned la = ldsA + (unsigned)buf * BUFB + ldsRow;
        const unsigned lb = ldsB + (unsigned)buf * BUFB + ldsRow;
        async_copy_b128(la,          XB, aOff);
        async_copy_b128(la + rowHop, XB, aOff + gRowHop);
        async_copy_b128(lb,          WB, bOff);
        async_copy_b128(lb + rowHop, WB, bOff + gRowHop);
        aOff += BK * 2;  // advance one K-step (64 B)
        bOff += BK * 2;
    };

    v8f acc[2][4];
#pragma unroll
    for (int i = 0; i < 2; ++i)
#pragma unroll
        for (int j = 0; j < 4; ++j) acc[i][j] = {};

    // Per-wave fragment element offsets within one buffer.
    const int aFrag0 = (wm * 32) * LDK;
    const int bFrag0 = (wn * 64) * LDK;

    auto compute = [&](int buf) {
        const __bf16* Ab = &As[0][0] + buf * BUFE + aFrag0;
        const __bf16* Bb = &Bs[0][0] + buf * BUFE + bFrag0;
        v16bf af[2], bfr[4];
#pragma unroll
        for (int i = 0; i < 2; ++i) af[i] = load_frag(Ab + (16 * i) * LDK, lane);
#pragma unroll
        for (int j = 0; j < 4; ++j) bfr[j] = load_frag(Bb + (16 * j) * LDK, lane);
#pragma unroll
        for (int i = 0; i < 2; ++i)
#pragma unroll
            for (int j = 0; j < 4; ++j)
                acc[i][j] = __builtin_amdgcn_wmma_f32_16x16x32_bf16(
                    false, af[i], false, bfr[j], (short)0, acc[i][j], false,
                    false);
    };

    // ---- prologue: stages 0 and 1 in flight ----
    issue_stage(0);
    issue_stage(1);
    wait_async_le4();  // stage 0 landed (stage 1 may still fly)
    __syncthreads();

    // ---- steady state: issue stage c+2, compute stage c ----
    int bufC = 0;  // buffer being computed
    int bufI = 2;  // buffer being filled
#pragma unroll 1
    for (int it = 0; it < (K_DIM / BK) - 2; ++it) {  // 126 iterations
        issue_stage(bufI);
        compute(bufC);
        wait_async_le4();  // stage it+1 landed; stage it+2 may still fly
        __syncthreads();
        bufC = (bufC == 2) ? 0 : bufC + 1;
        bufI = (bufI == 2) ? 0 : bufI + 1;
    }

    // ---- drain: stages 126, 127 ----
    compute(bufC);
    wait_async_0();
    __syncthreads();
    bufC = (bufC == 2) ? 0 : bufC + 1;
    compute(bufC);

    // ---- epilogue (non-temporal: out is write-once) ----
    const int colLane = lane & 15;
    const int rowOff  = (lane >> 4) * 8;
#pragma unroll
    for (int i = 0; i < 2; ++i) {
#pragma unroll
        for (int j = 0; j < 4; ++j) {
            const int col  = nBase + wn * 64 + j * 16 + colLane;
            const int row0 = mBase + wm * 32 + i * 16 + rowOff;
            const float bv = bias[col];
#pragma unroll
            for (int v = 0; v < 8; ++v)
                __builtin_nontemporal_store(
                    acc[i][j][v] + bv,
                    out + (size_t)(row0 + v) * N_DIM + col);
        }
    }
}

// ============================================================
// GEMM (fused fallback): fp32 in, convert/binarize in-kernel.
// ============================================================
__global__ __launch_bounds__(256)
void gemm_bf16_fused(const float* __restrict__ X, const float* __restrict__ W,
                     const float* __restrict__ bias, float* __restrict__ out,
                     const float* __restrict__ st) {
    __shared__ __bf16 As[2][BUFE];
    __shared__ __bf16 Bs[2][BUFE];

    const int tid  = threadIdx.x;
    const int lane = tid & 31;
    const int wave = tid >> 5;
    const int wm   = wave >> 1;
    const int wn   = wave & 1;

    const int nBase = blockIdx.x * 128;
    const int mBase = blockIdx.y * 128;

    const float lo = st[0], hi = st[1], scale = st[2];

    const int lrow = tid >> 3;
    const int lcol = (tid & 7) << 2;
    const size_t aOff = (size_t)(mBase + lrow) * K_DIM + lcol;
    const size_t bOff = (size_t)(nBase + lrow) * K_DIM + lcol;

    v8f acc[2][4];
#pragma unroll
    for (int i = 0; i < 2; ++i)
#pragma unroll
        for (int j = 0; j < 4; ++j) acc[i][j] = {};

    float4 ax[4], bw[4];

    auto g_load = [&](int k0) {
#pragma unroll
        for (int r = 0; r < 4; ++r) {
            ax[r] = *(const float4*)(X + aOff + (size_t)(32 * r) * K_DIM + k0);
            bw[r] = *(const float4*)(W + bOff + (size_t)(32 * r) * K_DIM + k0);
        }
    };
    auto bin = [&](float v) { return (v < lo || v > hi) ? v : v * scale; };
    auto stage = [&](int buf) {
#pragma unroll
        for (int r = 0; r < 4; ++r) {
            const int row = lrow + 32 * r;
            v4bf pa;
            pa[0] = (__bf16)ax[r].x;
            pa[1] = (__bf16)ax[r].y;
            pa[2] = (__bf16)ax[r].z;
            pa[3] = (__bf16)ax[r].w;
            *(v4bf*)(&As[buf][row * LDK + lcol]) = pa;
            v4bf pb;
            pb[0] = (__bf16)bin(bw[r].x);
            pb[1] = (__bf16)bin(bw[r].y);
            pb[2] = (__bf16)bin(bw[r].z);
            pb[3] = (__bf16)bin(bw[r].w);
            *(v4bf*)(&Bs[buf][row * LDK + lcol]) = pb;
        }
    };
    auto compute = [&](int buf) {
        v16bf af[2], bfr[4];
#pragma unroll
        for (int i = 0; i < 2; ++i)
            af[i] = load_frag(&As[buf][(wm * 32 + 16 * i) * LDK], lane);
#pragma unroll
        for (int j = 0; j < 4; ++j)
            bfr[j] = load_frag(&Bs[buf][(wn * 64 + 16 * j) * LDK], lane);
#pragma unroll
        for (int i = 0; i < 2; ++i)
#pragma unroll
            for (int j = 0; j < 4; ++j)
                acc[i][j] = __builtin_amdgcn_wmma_f32_16x16x32_bf16(
                    false, af[i], false, bfr[j], (short)0, acc[i][j], false,
                    false);
    };

    g_load(0);
    stage(0);
    __syncthreads();
    int buf = 0;
    for (int k0 = BK; k0 < K_DIM; k0 += BK) {
        g_load(k0);
        compute(buf);
        stage(buf ^ 1);
        __syncthreads();
        buf ^= 1;
    }
    compute(buf);

    const int colLane = lane & 15;
    const int rowOff  = (lane >> 4) * 8;
#pragma unroll
    for (int i = 0; i < 2; ++i) {
#pragma unroll
        for (int j = 0; j < 4; ++j) {
            const int col  = nBase + wn * 64 + j * 16 + colLane;
            const int row0 = mBase + wm * 32 + i * 16 + rowOff;
            const float bv = bias[col];
#pragma unroll
            for (int v = 0; v < 8; ++v)
                __builtin_nontemporal_store(
                    acc[i][j][v] + bv,
                    out + (size_t)(row0 + v) * N_DIM + col);
        }
    }
}

// ============================================================
extern "C" void kernel_launch(void* const* d_in, const int* in_sizes, int n_in,
                              void* d_out, int out_size, void* d_ws,
                              size_t ws_size, hipStream_t stream) {
    (void)in_sizes; (void)n_in; (void)out_size;
    const float* x    = (const float*)d_in[0];
    const float* w    = (const float*)d_in[1];
    const float* bias = (const float*)d_in[2];
    float*  out = (float*)d_out;
    double* wsd = (double*)d_ws;
    const float* st = (const float*)(wsd + WS_F_OFF);

    stats_partial1<<<RED_BLOCKS, RED_THREADS, 0, stream>>>(w, wsd);
    stats_finalize1<<<1, RED_THREADS, 0, stream>>>(wsd);
    stats_partial2<<<RED_BLOCKS, RED_THREADS, 0, stream>>>(w, wsd);
    stats_finalize2<<<1, RED_THREADS, 0, stream>>>(wsd);

    const size_t WBF_OFF = (size_t)1 << 20;                       // 1 MB
    const size_t XBF_OFF = WBF_OFF + (size_t)N_DIM * K_DIM * 2;   // +32 MB
    const size_t NEED    = XBF_OFF + (size_t)M_DIM * K_DIM * 2;   // +64 MB

    dim3 grid(N_DIM / 128, M_DIM / 128);  // 32 x 64

    if (ws_size >= NEED) {
        __bf16* wb = (__bf16*)((char*)d_ws + WBF_OFF);
        __bf16* xb = (__bf16*)((char*)d_ws + XBF_OFF);
        cvt_w_bf16<<<NW / 4 / 256, 256, 0, stream>>>(w, wb, st);
        cvt_x_bf16<<<(size_t)M_DIM * K_DIM / 4 / 256, 256, 0, stream>>>(x, xb);
        gemm_bf16_async<<<grid, 256, 0, stream>>>(xb, wb, bias, out);
    } else {
        gemm_bf16_fused<<<grid, 256, 0, stream>>>(x, w, bias, out, st);
    }
}
